// Conv2DRNNGenerator_48498770706828
// MI455X (gfx1250) — matrix operand order repaired and checked by reading, not compiled, not run on hardware
//
#include <hip/hip_runtime.h>
#include <hip/hip_bf16.h>

typedef __attribute__((ext_vector_type(16))) __bf16 v16bf;
typedef __attribute__((ext_vector_type(8)))  float  v8f;

#define BB 64
#define TT 64
#define FF 256
#define CC 64
#define KK 9
#define GC (4*CC)     // 256 gate channels
#define KDIM (KK*CC)  // 576 = im2col K

__device__ __forceinline__ __bf16 f32_to_bf16(float f) {
  union { float f; unsigned u; } v; v.f = f;
  unsigned r = v.u + 0x7FFFu + ((v.u >> 16) & 1u);   // round-to-nearest-even
  unsigned short h = (unsigned short)(r >> 16);
  __bf16 o; __builtin_memcpy(&o, &h, 2); return o;
}

__device__ __forceinline__ float sigmoidf_(float x) {
  return 1.0f / (1.0f + __expf(-x));
}

// ---- Preprocessing: transpose Wh (9,64,256) fp32 -> WhT[co][576] bf16 ----
__global__ __launch_bounds__(256) void prep_weights(const float* __restrict__ Wh,
                                                    __bf16* __restrict__ WhT) {
  int i = blockIdx.x * blockDim.x + threadIdx.x;
  if (i < KDIM * GC) {
    int kci = i / GC, co = i % GC;
    WhT[(size_t)co * KDIM + kci] = f32_to_bf16(Wh[i]);
  }
}

// ---- Preprocessing: h0 fp32 -> bf16 state buffer, c0 fp32 copy ----
__global__ __launch_bounds__(256) void prep_state(const float* __restrict__ h0,
                                                  const float* __restrict__ c0,
                                                  __bf16* __restrict__ hb,
                                                  float* __restrict__ cb) {
  int i = blockIdx.x * blockDim.x + threadIdx.x;
  if (i < BB * FF * CC) {
    hb[i] = f32_to_bf16(h0[i]);
    cb[i] = c0[i];
  }
}

// ---- One ConvLSTM timestep, fully fused ----
// grid: (16 frame-tiles, 64 batches), block: 256 threads = 8 waves
__global__ __launch_bounds__(256) void convlstm_step(
    const __bf16* __restrict__ h_in,   // (B,F,C) bf16
    const float*  __restrict__ c_in,   // (B,F,C) f32
    __bf16* __restrict__ h_out,
    float*  __restrict__ c_out,
    const __bf16* __restrict__ WhT,    // [256][576] bf16
    const float*  __restrict__ Wx,     // (9,1,256) f32
    const float*  __restrict__ bias,   // (256,) f32
    const float*  __restrict__ z,      // (B,T,F) f32
    float* __restrict__ out,           // (B, T*F) f32
    int t)
{
  __shared__ __bf16 sh_h[24][CC];       // halo'd h tile (A operand source)
  __shared__ float  sh_z[24];
  __shared__ float  sh_gates[16][GC];   // fused gate buffer
  __shared__ float  sh_red[16][16];

  const int ft   = blockIdx.x;          // frame tile 0..15
  const int b    = blockIdx.y;          // batch 0..63
  const int tid  = threadIdx.x;
  const int lane = tid & 31;
  const int wave = tid >> 5;
  const int f0   = ft * 16;

  // Stage h rows [f0-4, f0+20) as bf16 (zero halo), plus z window
  for (int i = tid; i < 24 * CC; i += 256) {
    int r = i >> 6, ci = i & 63;
    int f = f0 - 4 + r;
    __bf16 v; { unsigned short zz = 0; __builtin_memcpy(&v, &zz, 2); }
    if (f >= 0 && f < FF) v = h_in[((size_t)b * FF + f) * CC + ci];
    sh_h[r][ci] = v;
  }
  if (tid < 24) {
    int f = f0 - 4 + tid;
    sh_z[tid] = (f >= 0 && f < FF) ? z[((size_t)b * TT + t) * FF + f] : 0.0f;
  }
  __syncthreads();

  // ---- GEMM: gates[16 x 256] += im2col(h)[16 x 576] @ Wh[576 x 256] ----
  const int halfsel = lane >> 4;        // 0: K 0..15 (A), M 0..7 (C)
  const int mrow    = lane & 15;        // A row
  const int ncol    = lane & 15;        // B/C column within 16-tile
  const int n0      = wave * 32;
  const int co0     = n0 + ncol;
  const int co1     = n0 + 16 + ncol;

  v8f acc0 = {}; v8f acc1 = {};
  const __bf16* wr0 = WhT + (size_t)co0 * KDIM;
  const __bf16* wr1 = WhT + (size_t)co1 * KDIM;

  for (int kc = 0; kc < KDIM / 32; ++kc) {
    // A fragment: 16x32 bf16; K-chunk kc lies inside kernel-tap kc>>1
    v16bf a;
    const __bf16* s0 = &sh_h[mrow + (kc >> 1)][(kc & 1) * 32 + halfsel * 16];
    #pragma unroll
    for (int j = 0; j < 8; ++j) a[j] = s0[j];
    #pragma unroll
    for (int j = 0; j < 8; ++j) a[8 + j] = s0[8 + j];
    // B fragments: contiguous 32B per lane from pre-transposed weights
    v16bf b0 = *(const v16bf*)(wr0 + kc * 32 + halfsel * 16);
    v16bf b1 = *(const v16bf*)(wr1 + kc * 32 + halfsel * 16);
    acc0 = __builtin_amdgcn_wmma_f32_16x16x32_bf16(false, a, false, b0,
                                                   (short)0, acc0, false, false);
    acc1 = __builtin_amdgcn_wmma_f32_16x16x32_bf16(false, a, false, b1,
                                                   (short)0, acc1, false, false);
  }

  // ---- Epilogue: + z-conv (Cin=1) + bias, spill gates to LDS ----
  {
    float wx9[9];
    #pragma unroll
    for (int k = 0; k < KK; ++k) wx9[k] = Wx[k * GC + co0];
    float bs = bias[co0];
    #pragma unroll
    for (int r = 0; r < 8; ++r) {
      int m = halfsel * 8 + r;
      float g = acc0[r] + bs;
      #pragma unroll
      for (int k = 0; k < KK; ++k) g += sh_z[m + k] * wx9[k];
      sh_gates[m][co0] = g;
    }
    #pragma unroll
    for (int k = 0; k < KK; ++k) wx9[k] = Wx[k * GC + co1];
    bs = bias[co1];
    #pragma unroll
    for (int r = 0; r < 8; ++r) {
      int m = halfsel * 8 + r;
      float g = acc1[r] + bs;
      #pragma unroll
      for (int k = 0; k < KK; ++k) g += sh_z[m + k] * wx9[k];
      sh_gates[m][co1] = g;
    }
  }
  __syncthreads();

  // ---- Pointwise LSTM update + fused channel-mean tanh output ----
  {
    const int m  = tid >> 4;            // frame row 0..15
    const int cg = (tid & 15) * 4;      // 4 channels per thread
    const int f  = f0 + m;
    const size_t base = ((size_t)b * FF + f) * CC;
    float psum = 0.0f;
    #pragma unroll
    for (int q = 0; q < 4; ++q) {
      int ch = cg + q;
      float ig = sh_gates[m][ch];
      float jg = sh_gates[m][CC + ch];
      float fg = sh_gates[m][2 * CC + ch];
      float og = sh_gates[m][3 * CC + ch];
      float cold = c_in[base + ch];
      float cnew = cold * sigmoidf_(fg + 1.0f) + sigmoidf_(ig) * tanhf(jg);
      float hnew = tanhf(cnew) * sigmoidf_(og);
      c_out[base + ch] = cnew;
      h_out[base + ch] = f32_to_bf16(hnew);
      psum += hnew;
    }
    sh_red[m][tid & 15] = psum;
  }
  __syncthreads();
  if (tid < 16) {
    float s = 0.0f;
    #pragma unroll
    for (int q = 0; q < 16; ++q) s += sh_red[tid][q];
    out[(size_t)b * (TT * FF) + (size_t)t * FF + (f0 + tid)] =
        tanhf(s * (1.0f / CC));
  }
}

extern "C" void kernel_launch(void* const* d_in, const int* in_sizes, int n_in,
                              void* d_out, int out_size, void* d_ws, size_t ws_size,
                              hipStream_t stream) {
  const float* z    = (const float*)d_in[0];  // (B,T,F,1)
  const float* h0   = (const float*)d_in[1];  // (B,F,C)
  const float* c0   = (const float*)d_in[2];  // (B,F,C)
  const float* Wx   = (const float*)d_in[3];  // (9,1,256)
  const float* Wh   = (const float*)d_in[4];  // (9,64,256)
  const float* bias = (const float*)d_in[5];  // (256,)
  float* out = (float*)d_out;

  char* ws = (char*)d_ws;
  const size_t whT_bytes = (size_t)GC * KDIM * sizeof(__bf16);      // 294912
  const size_t h_bytes   = (size_t)BB * FF * CC * sizeof(__bf16);   // 2 MiB
  const size_t c_bytes   = (size_t)BB * FF * CC * sizeof(float);    // 4 MiB
  __bf16* WhT   = (__bf16*)ws;
  __bf16* hbuf0 = (__bf16*)(ws + whT_bytes);
  __bf16* hbuf1 = (__bf16*)(ws + whT_bytes + h_bytes);
  float*  cbuf0 = (float*)(ws + whT_bytes + 2 * h_bytes);
  float*  cbuf1 = (float*)(ws + whT_bytes + 2 * h_bytes + c_bytes);

  prep_weights<<<(KDIM * GC + 255) / 256, 256, 0, stream>>>(Wh, WhT);
  prep_state<<<(BB * FF * CC + 255) / 256, 256, 0, stream>>>(h0, c0, hbuf0, cbuf0);

  __bf16* hin = hbuf0; __bf16* hout = hbuf1;
  float*  cin = cbuf0; float*  cout = cbuf1;
  for (int t = 0; t < TT; ++t) {
    convlstm_step<<<dim3(16, 64), 256, 0, stream>>>(
        hin, cin, hout, cout, WhT, Wx, bias, z, out, t);
    { __bf16* tmp = hin; hin = hout; hout = tmp; }
    { float*  tmp = cin; cin = cout; cout = tmp; }
  }
  (void)in_sizes; (void)n_in; (void)out_size; (void)ws_size;
}
